// CrossAttention2D_56736517980251
// MI455X (gfx1250) — matrix operand order repaired and checked by reading, not compiled, not run on hardware
//
#include <hip/hip_runtime.h>
#include <stdint.h>

// ---------------------------------------------------------------------------
// CDNA5 (gfx1250) fused cross-attention:
//   Q/K/V 1x1-conv projections (WMMA f16) + positional encoding
//   flash attention with online softmax (WMMA f16, TDM tile DMA)
//   output 1x1-conv projection (WMMA f16)
// Problem: B=4, C=D=256, H=W=64 -> N=4096
// ---------------------------------------------------------------------------

typedef __attribute__((ext_vector_type(16))) _Float16 v16h;
typedef __attribute__((ext_vector_type(8)))  float    v8f;
typedef __attribute__((ext_vector_type(4)))  uint32_t v4u32;
typedef __attribute__((ext_vector_type(8)))  int32_t  v8i32;
typedef __attribute__((ext_vector_type(4)))  int32_t  v4i32;
typedef __attribute__((ext_vector_type(4)))  uint32_t u32x4;

union F16x16 { v16h v; u32x4 q[2]; };

#define LDS_OFF(p) ((uint32_t)(uintptr_t)(p))

__device__ __forceinline__ v8f v8f_zero() {
  v8f z;
#pragma unroll
  for (int i = 0; i < 8; ++i) z[i] = 0.0f;
  return z;
}

__device__ __forceinline__ v8f wmma16(v16h a, v16h b, v8f c) {
  // D = A(16x32 f16) * B(32x16 f16) + C(16x16 f32)
  return __builtin_amdgcn_wmma_f32_16x16x32_f16(false, a, false, b, (short)0, c,
                                                false, false);
}

// A-fragment (16x32 f16), source row-major with leading dim `ld`.
// ISA layout: lane L -> row M = row0 + (L&15); lanes 0-15 hold K = k0+[0..7],
// k0+[16..23]; lanes 16-31 hold K = k0+[8..15], k0+[24..31].
__device__ __forceinline__ v16h load_a_frag(const _Float16* m, int ld, int row0,
                                            int k0) {
  const int lane = threadIdx.x & 31;
  const _Float16* p = m + (row0 + (lane & 15)) * ld + k0 + ((lane >> 4) << 3);
  F16x16 f;
  f.q[0] = *reinterpret_cast<const u32x4*>(p);       // K chunk lo (16B)
  f.q[1] = *reinterpret_cast<const u32x4*>(p + 16);  // K chunk hi (16B)
  return f.v;
}

// B-fragment (32x16 f16) gathered from B^T stored row-major (bt[n][k], lead ld).
// ISA layout: lane L -> column N = n0 + (L&15); lanes 0-15 hold K=k0+[0..7],
// k0+[16..23]; lanes 16-31 hold K=k0+[8..15], k0+[24..31].
__device__ __forceinline__ v16h load_bt_frag(const _Float16* bt, int ld, int n0,
                                             int k0) {
  const int lane = threadIdx.x & 31;
  const _Float16* p = bt + (n0 + (lane & 15)) * ld + k0 + ((lane >> 4) << 3);
  F16x16 f;
  f.q[0] = *reinterpret_cast<const u32x4*>(p);
  f.q[1] = *reinterpret_cast<const u32x4*>(p + 16);
  return f.v;
}

// ---------------------------------------------------------------------------
// Tensor Data Mover: 2D tile (tile_w x tile_h elements, 2B/elem) from a
// row-major tensor (row pitch `pitch_elems`) into LDS at byte offset lds_off.
// Descriptor packing per CDNA5 ISA 8.3/8.4 (type=2, count=1, data_size=1).
// ---------------------------------------------------------------------------
__device__ __forceinline__ void tdm_load_2d(uint32_t lds_off, const void* gptr,
                                            uint32_t tile_w, uint32_t tile_h,
                                            uint32_t tensor_w, uint32_t tensor_h,
                                            uint32_t pitch_elems) {
  const uint64_t ga = (uint64_t)(uintptr_t)gptr;
  v4u32 g0;
  g0[0] = 1u;                    // count=1 valid descriptor, no gather
  g0[1] = lds_off;               // lds_addr (bytes)
  g0[2] = (uint32_t)ga;          // global_addr[31:0]
  g0[3] = ((uint32_t)(ga >> 32) & 0x01FFFFFFu) | (2u << 30);  // addr[56:32]|type=2
  v8i32 g1;
  g1[0] = (int32_t)(1u << 16);   // workgroup_mask=0, data_size=1 (2 bytes)
  g1[1] = (int32_t)((tensor_w & 0xFFFFu) << 16);                    // dim0 lo
  g1[2] = (int32_t)((tensor_w >> 16) | ((tensor_h & 0xFFFFu) << 16));
  g1[3] = (int32_t)((tensor_h >> 16) | (tile_w << 16));             // tile_dim0
  g1[4] = (int32_t)(tile_h & 0xFFFFu);                              // tile_dim1
  g1[5] = (int32_t)pitch_elems;  // tensor_dim0_stride[31:0]
  g1[6] = 0;
  g1[7] = 0;
  v4i32 z4; z4[0] = z4[1] = z4[2] = z4[3] = 0;
#if defined(__clang_major__) && (__clang_major__ >= 23)
  v8i32 z8;
#pragma unroll
  for (int i = 0; i < 8; ++i) z8[i] = 0;
  __builtin_amdgcn_tensor_load_to_lds(g0, g1, z4, z4, z8, 0);
#else
  __builtin_amdgcn_tensor_load_to_lds(g0, g1, z4, z4, 0);
#endif
}

// Positional encoding value pe[d, h, w] for D=256, H=W=64 (ch_half=128).
__device__ __forceinline__ float pe_val(int d, int n) {
  const int h = n >> 6, w = n & 63;
  const int t = (d < 128) ? w : h;
  const int j = (d < 128) ? d : d - 128;
  const int jj = (j < 64) ? j : (j - 64);
  // inv[i] = 10000^(-i/64) = exp(-i * ln(10000)/64)
  const float arg = (float)t * __expf(-0.14391156831212787f * (float)jj);
  return (j < 64) ? __sinf(arg) : __cosf(arg);
}

// ---------------------------------------------------------------------------
// Kernel 1: f32 -> f16 weight conversion (weights [out][in] are already B^T).
// ---------------------------------------------------------------------------
__global__ __launch_bounds__(256) void wconv_kernel(const float* __restrict__ src,
                                                    _Float16* __restrict__ dst,
                                                    int n) {
  const int i = blockIdx.x * 256 + threadIdx.x;
  if (i < n) dst[i] = (_Float16)src[i];
}

// ---------------------------------------------------------------------------
// Kernel 2: QKV projection.  One block = one batch b and 64 spatial rows.
// X: [B,256,N] f32 (Y or S).  Wh: [256][256] f16.  mode 0=Q(+PE) 1=K(+PE)
// 2=V stored transposed [B,D,N] so the attention P*V B-fragments are
// contiguous in LDS.
// ---------------------------------------------------------------------------
#define XPITCH 264  // padded LDS pitch (multiple of 8 keeps 16B alignment)

__global__ __launch_bounds__(256) void qkv_proj_kernel(
    const float* __restrict__ X, const _Float16* __restrict__ Wh,
    const float* __restrict__ bias, _Float16* __restrict__ Out, int mode) {
  __shared__ __align__(16) _Float16 lds_x[64 * XPITCH];  // X tile [64 n][256 c]
  __shared__ __align__(16) _Float16 lds_w[256 * 32];     // W^T chunk [256 d][32 c]

  const int tid = threadIdx.x, wv = tid >> 5, lane = tid & 31;
  const int b = blockIdx.x >> 6;
  const int n0 = (blockIdx.x & 63) * 64;
  const int N = 4096, C = 256, D = 256;

  // Stage + transpose + convert X tile: global [c][n] -> LDS [n][c] f16.
  {
    const int nl = tid & 63;
    const size_t xb = (size_t)b * C * N + (size_t)(n0 + nl);
    for (int c = tid >> 6; c < C; c += 4)
      lds_x[nl * XPITCH + c] = (_Float16)X[xb + (size_t)c * N];
  }

  const int i16 = (wv >> 1) * 16;  // row-tile base within 64-row tile
  const int ch = (wv & 1) * 128;   // column half
  v8f acc[8];
#pragma unroll
  for (int t = 0; t < 8; ++t) acc[t] = v8f_zero();

  for (int ks = 0; ks < 8; ++ks) {  // k over C=256 in chunks of 32
    __syncthreads();
    if (wv == 0) {  // TDM: stream 32-wide weight column chunk into LDS
      tdm_load_2d(LDS_OFF(lds_w), Wh + ks * 32, 32, 256, 256, 256, 256);
      __builtin_amdgcn_s_wait_tensorcnt(0);
    }
    __syncthreads();
    const v16h a = load_a_frag(lds_x, XPITCH, i16, ks * 32);
#pragma unroll
    for (int t = 0; t < 8; ++t) {
      const v16h bt = load_bt_frag(lds_w, 32, ch + t * 16, 0);
      acc[t] = wmma16(a, bt, acc[t]);
    }
  }

  // Epilogue: bias (+ PE for Q/K), convert to f16, store.
  const int cn = lane & 15;
  const int r8 = (lane >> 4) << 3;
#pragma unroll
  for (int t = 0; t < 8; ++t) {
    const int d = ch + t * 16 + cn;
    const float bb = bias[d];
#pragma unroll
    for (int r = 0; r < 8; ++r) {
      const int n = n0 + i16 + r8 + r;
      float v = acc[t][r] + bb;
      if (mode < 2) {
        v += pe_val(d, n);
        Out[((size_t)b * N + n) * D + d] = (_Float16)v;   // [B,N,D]
      } else {
        Out[((size_t)b * D + d) * N + n] = (_Float16)v;   // [B,D,N] (V^T)
      }
    }
  }
}

// ---------------------------------------------------------------------------
// Kernel 3: flash attention.  Block = 32 query rows of one batch; streams
// 64 chunks of 64 keys.  Q fragments are hoisted into registers (invariant
// over the key loop).  K chunk and V chunk share one 32KB LDS buffer; the V
// TDM load (wave 0) overlaps with the online-softmax update (waves 4-7).
// ---------------------------------------------------------------------------
__global__ __launch_bounds__(256) void flash_attn_kernel(
    const _Float16* __restrict__ Qf,  // [B,N,D]
    const _Float16* __restrict__ Kf,  // [B,N,D]
    const _Float16* __restrict__ Vt,  // [B,D,N]
    _Float16* __restrict__ Z) {       // [B,N,D]
  __shared__ __align__(16) _Float16 lds_q[32 * 256];
  __shared__ __align__(16) _Float16 lds_kv[64 * 256];  // K [64][256] or V [256][64]
  __shared__ __align__(16) float lds_s[32 * 64];
  __shared__ __align__(16) _Float16 lds_p[32 * 64];
  __shared__ float row_m[32], row_l[32], row_c[32];

  const int tid = threadIdx.x, wv = tid >> 5, lane = tid & 31;
  const int b = blockIdx.x >> 7;
  const int q0 = (blockIdx.x & 127) * 32;
  const int N = 4096, D = 256;
  const size_t base = (size_t)b * N * D;

  if (wv == 0) {  // TDM: Q tile [32][256]
    tdm_load_2d(LDS_OFF(lds_q), Qf + base + (size_t)q0 * D, 256, 32, 256, 4096, 256);
    __builtin_amdgcn_s_wait_tensorcnt(0);
  }
  if (tid < 32) { row_m[tid] = -3.0e38f; row_l[tid] = 0.0f; }
  __syncthreads();

  const int c0 = wv * 32;                             // this wave's output cols
  const int si = (wv >> 2) * 16, sj = (wv & 3) * 16;  // this wave's S tile
  // Hoist Q A-fragments: invariant across all 64 key chunks (64 VGPRs).
  v16h qf[8];
#pragma unroll
  for (int ks = 0; ks < 8; ++ks) qf[ks] = load_a_frag(lds_q, 256, si, ks * 32);

  v8f o[4];
#pragma unroll
  for (int t = 0; t < 4; ++t) o[t] = v8f_zero();
  const float scale = 0.0625f;  // 256^-0.5

  for (int kc = 0; kc < 64; ++kc) {
    __syncthreads();  // previous P*V reads of lds_kv complete
    if (wv == 0) {    // TDM: K chunk [64][256]
      tdm_load_2d(LDS_OFF(lds_kv), Kf + base + (size_t)(kc * 64) * D, 256, 64,
                  256, 4096, 256);
      __builtin_amdgcn_s_wait_tensorcnt(0);
    }
    __syncthreads();

    // S = (Q K^T) tile (16x16 per wave), scaled, written to LDS.
    v8f s = v8f_zero();
#pragma unroll
    for (int ks = 0; ks < 8; ++ks) {
      const v16h bt = load_bt_frag(lds_kv, 256, sj, ks * 32);
      s = wmma16(qf[ks], bt, s);
    }
    {
      const int col = sj + (lane & 15);
      const int rb = si + ((lane >> 4) << 3);
#pragma unroll
      for (int r = 0; r < 8; ++r) lds_s[(rb + r) * 64 + col] = s[r] * scale;
    }
    __syncthreads();  // S visible; all waves done reading K

    // Wave 0: DMA the V chunk (overwrites lds_kv, K fully consumed) ...
    if (wv == 0) {
      tdm_load_2d(LDS_OFF(lds_kv), Vt + (size_t)b * D * N + (size_t)(kc * 64),
                  64, 256, 4096, 256, 4096);
      __builtin_amdgcn_s_wait_tensorcnt(0);
    }
    // ... while waves 4-7 run the online softmax (4 lanes per row, 16 cols
    // each, cross-lane reduction via shuffles).
    if (wv >= 4) {
      const int r = (tid - 128) >> 2;  // 0..31
      const int sub = tid & 3;
      const float* srow = lds_s + r * 64 + sub * 16;
      float mn = -3.0e38f;
#pragma unroll
      for (int c = 0; c < 16; ++c) mn = fmaxf(mn, srow[c]);
      mn = fmaxf(mn, __shfl_xor(mn, 1));
      mn = fmaxf(mn, __shfl_xor(mn, 2));
      const float mo = row_m[r];
      mn = fmaxf(mn, mo);
      float ps = 0.0f;
      _Float16* prow = lds_p + r * 64 + sub * 16;
#pragma unroll
      for (int c = 0; c < 16; ++c) {
        const float p = __expf(srow[c] - mn);
        ps += p;
        prow[c] = (_Float16)p;
      }
      ps += __shfl_xor(ps, 1);
      ps += __shfl_xor(ps, 2);
      if (sub == 0) {
        const float corr = __expf(mo - mn);
        row_m[r] = mn;
        row_l[r] = row_l[r] * corr + ps;
        row_c[r] = corr;
      }
    }
    __syncthreads();

    // Rescale accumulators, then O += P @ V (V chunk is [d][key] in LDS).
#pragma unroll
    for (int t = 0; t < 4; ++t) {
      const int rb = (t >> 1) * 16 + ((lane >> 4) << 3);
#pragma unroll
      for (int r = 0; r < 8; ++r) o[t][r] *= row_c[rb + r];
    }
#pragma unroll
    for (int t = 0; t < 4; ++t) {
      const int ti = (t >> 1) * 16;
      const int dj = c0 + (t & 1) * 16;
#pragma unroll
      for (int kk = 0; kk < 2; ++kk) {
        const v16h a = load_a_frag(lds_p, 64, ti, kk * 32);
        const v16h bt = load_bt_frag(lds_kv, 64, dj, kk * 32);
        o[t] = wmma16(a, bt, o[t]);
      }
    }
  }

  __syncthreads();
#pragma unroll
  for (int t = 0; t < 4; ++t) {
    const int d = c0 + (t & 1) * 16 + (lane & 15);
    const int rb = (t >> 1) * 16 + ((lane >> 4) << 3);
#pragma unroll
    for (int r = 0; r < 8; ++r) {
      const int row = rb + r;
      Z[base + (size_t)(q0 + row) * D + d] = (_Float16)(o[t][r] / row_l[row]);
    }
  }
}

// ---------------------------------------------------------------------------
// Kernel 4: output projection.  out[b][c][n] = sum_d Wo[c][d]*Z[b][n][d]+bo[c].
// ---------------------------------------------------------------------------
__global__ __launch_bounds__(256) void out_proj_kernel(
    const _Float16* __restrict__ Z,   // [B,N,D]
    const _Float16* __restrict__ WoH, // [256 c][256 d] f16 (== B^T)
    const float* __restrict__ bo, float* __restrict__ Out) {  // [B,256,N]
  __shared__ __align__(16) _Float16 lds_z[64 * 256];
  __shared__ __align__(16) _Float16 lds_w[256 * 32];

  const int tid = threadIdx.x, wv = tid >> 5, lane = tid & 31;
  const int b = blockIdx.x >> 6;
  const int n0 = (blockIdx.x & 63) * 64;
  const int N = 4096, D = 256;

  if (wv == 0) {  // TDM: Z tile [64][256]
    tdm_load_2d(LDS_OFF(lds_z), Z + ((size_t)b * N + n0) * D, 256, 64, 256, 4096,
                256);
    __builtin_amdgcn_s_wait_tensorcnt(0);
  }

  const int i16 = (wv >> 1) * 16;
  const int ch = (wv & 1) * 128;
  v8f acc[8];
#pragma unroll
  for (int t = 0; t < 8; ++t) acc[t] = v8f_zero();

  for (int ks = 0; ks < 8; ++ks) {
    __syncthreads();
    if (wv == 0) {
      tdm_load_2d(LDS_OFF(lds_w), WoH + ks * 32, 32, 256, 256, 256, 256);
      __builtin_amdgcn_s_wait_tensorcnt(0);
    }
    __syncthreads();
    const v16h a = load_a_frag(lds_z, 256, i16, ks * 32);
#pragma unroll
    for (int t = 0; t < 8; ++t) {
      const v16h bt = load_bt_frag(lds_w, 32, ch + t * 16, 0);
      acc[t] = wmma16(a, bt, acc[t]);
    }
  }

  const int cn = lane & 15;
  const int r8 = (lane >> 4) << 3;
#pragma unroll
  for (int t = 0; t < 8; ++t) {
    const int c = ch + t * 16 + cn;
    const float bb = bo[c];
#pragma unroll
    for (int r = 0; r < 8; ++r) {
      const int n = n0 + i16 + r8 + r;
      Out[((size_t)b * 256 + c) * N + n] = acc[t][r] + bb;
    }
  }
}

// ---------------------------------------------------------------------------
// Host launch.  Workspace layout (bytes):
//   [0, 512K):  WqH/WkH/WvH/WoH f16 copies (128K each)
//   then Qf, Kf, Vt(=V^T), Z: 8MB each (f16 [4,4096,256])  -> ~32.5 MB total
// ---------------------------------------------------------------------------
extern "C" void kernel_launch(void* const* d_in, const int* in_sizes, int n_in,
                              void* d_out, int out_size, void* d_ws,
                              size_t ws_size, hipStream_t stream) {
  (void)in_sizes; (void)n_in; (void)out_size; (void)ws_size;
  const float* Y  = (const float*)d_in[0];
  const float* S  = (const float*)d_in[1];
  const float* Wq = (const float*)d_in[2];
  const float* bq = (const float*)d_in[3];
  const float* Wk = (const float*)d_in[4];
  const float* bk = (const float*)d_in[5];
  const float* Wv = (const float*)d_in[6];
  const float* bv = (const float*)d_in[7];
  const float* Wo = (const float*)d_in[8];
  const float* bo = (const float*)d_in[9];
  float* out = (float*)d_out;
  char* ws = (char*)d_ws;

  _Float16* WqH = (_Float16*)(ws);
  _Float16* WkH = (_Float16*)(ws + (128ull << 10));
  _Float16* WvH = (_Float16*)(ws + (256ull << 10));
  _Float16* WoH = (_Float16*)(ws + (384ull << 10));
  _Float16* Qf  = (_Float16*)(ws + (512ull << 10));
  _Float16* Kf  = (_Float16*)(ws + (512ull << 10) + (8ull << 20));
  _Float16* Vt  = (_Float16*)(ws + (512ull << 10) + (16ull << 20));
  _Float16* Zb  = (_Float16*)(ws + (512ull << 10) + (24ull << 20));

  wconv_kernel<<<256, 256, 0, stream>>>(Wq, WqH, 65536);
  wconv_kernel<<<256, 256, 0, stream>>>(Wk, WkH, 65536);
  wconv_kernel<<<256, 256, 0, stream>>>(Wv, WvH, 65536);
  wconv_kernel<<<256, 256, 0, stream>>>(Wo, WoH, 65536);

  qkv_proj_kernel<<<256, 256, 0, stream>>>(Y, WqH, bq, Qf, 0);
  qkv_proj_kernel<<<256, 256, 0, stream>>>(S, WkH, bk, Kf, 1);
  qkv_proj_kernel<<<256, 256, 0, stream>>>(S, WvH, bv, Vt, 2);

  flash_attn_kernel<<<512, 256, 0, stream>>>(Qf, Kf, Vt, Zb);

  out_proj_kernel<<<256, 256, 0, stream>>>(Zb, WoH, bo, out);
}
